// TemporalMessageBank_76836964926247
// MI455X (gfx1250) — matrix operand order repaired and checked by reading, not compiled, not run on hardware
//
#include <hip/hip_runtime.h>

// Temporal message-bank attention, fused single kernel for gfx1250 (MI455X).
// bf16 WMMA (v_wmma_f32_16x16x32_bf16) for Q/K/V/O projections AND for the
// attention logits (block-diagonal extraction of Q@K^T) and ctx (attn@V via
// transposed-V layout). fp32 accumulate everywhere; fused residual + LN.
// Native __bf16 casts so the backend emits hardware cvt ops.

#define DIM       128
#define BANK      8
#define TQ        16              // queries per workgroup
#define PROWS     (TQ * BANK)     // 128 gathered past rows per workgroup
#define NTHREADS  128             // 4 waves (wave32)
#define LN_EPS    1e-5f
#define QK_SCALE  0.08838834764831845f   // 128^-0.5

typedef __bf16 bf16;
typedef __attribute__((ext_vector_type(8)))  __bf16 v8bf;
typedef __attribute__((ext_vector_type(16))) __bf16 v16bf;
typedef __attribute__((ext_vector_type(8)))  float  v8f;

__device__ inline bf16 f2bf(float f) { return (bf16)f; }     // hw v_cvt (RNE)
__device__ inline float bf2f(bf16 b) { return (float)b; }    // hw extend

// A-fragment 16x32 bf16 (ISA 7.12.2): lanes 0-15 hold row M=lane, K {0-7,16-23};
// lanes 16-31 hold row M=lane-16, K {8-15,24-31}. Two contiguous 16B LDS loads.
__device__ inline v16bf frag_a(const bf16* base, int lane, int kc) {
    int row = lane & 15;
    int kh  = (lane >> 4) << 3;               // 0 or 8
    const bf16* p = base + row * DIM + kc * 32 + kh;
    v8bf lo = *(const v8bf*)(p);
    v8bf hi = *(const v8bf*)(p + 16);
    return __builtin_shufflevector(lo, hi, 0,1,2,3,4,5,6,7,8,9,10,11,12,13,14,15);
}

// B-fragment 32x16 bf16: lane = column n (mod 16); lanes 0-15 hold K {0-7,16-23},
// lanes 16-31 hold K {8-15,24-31}. B = M^T with M row-major, so a B column is a
// contiguous M row segment -> two 16B LDS loads.
__device__ inline v16bf frag_b(const bf16* Mbase, int lane, int kc) {
    int n  = lane & 15;
    int kb = kc * 32 + ((lane >> 4) << 3);    // 0 or 8 within the 32-chunk
    const bf16* p = Mbase + n * DIM + kb;
    v8bf lo = *(const v8bf*)(p);
    v8bf hi = *(const v8bf*)(p + 16);
    return __builtin_shufflevector(lo, hi, 0,1,2,3,4,5,6,7,8,9,10,11,12,13,14,15);
}

// Cooperative load of a 128x128 fp32 weight matrix -> bf16 LDS (row-major).
__device__ inline void load_w(const float* __restrict__ W, bf16* dst, int t) {
    #pragma unroll 4
    for (int i = 0; i < (DIM * DIM) / (NTHREADS * 4); ++i) {   // 32 iters
        int base = (i * NTHREADS + t) * 4;
        float4 v = *(const float4*)(W + base);
        dst[base + 0] = f2bf(v.x);
        dst[base + 1] = f2bf(v.y);
        dst[base + 2] = f2bf(v.z);
        dst[base + 3] = f2bf(v.w);
    }
}

// D[rows x 128] = A[rows x 128] @ W^T + bias   (rows = 16*TR), bf16 out to LDS.
// STORE_T=false: row-major D.  STORE_T=true: store D transposed (for V^T).
template <bool STORE_T>
__device__ inline void proj_mm(const bf16* A, const bf16* W,
                               const float* __restrict__ bias,
                               bf16* D, int TR, int w, int lane) {
    const int perWave = TR * 2;
    for (int q = 0; q < perWave; ++q) {
        int id = w * perWave + q;
        int ti = id >> 3;
        int tj = id & 7;
        float bv = bias[tj * 16 + (lane & 15)];
        v8f acc;
        #pragma unroll
        for (int j = 0; j < 8; ++j) acc[j] = bv;
        const bf16* Ab = A + ti * 16 * DIM;
        const bf16* Wb = W + tj * 16 * DIM;
        #pragma unroll
        for (int c = 0; c < 4; ++c) {
            v16bf a = frag_a(Ab, lane, c);
            v16bf b = frag_b(Wb, lane, c);
            acc = __builtin_amdgcn_wmma_f32_16x16x32_bf16(
                      false, a, false, b, (short)0, acc, false, false);
        }
        // C/D layout: lane -> column n, lane-half -> rows m0..m0+7
        int n  = lane & 15;
        int m0 = (lane >> 4) << 3;
        if (STORE_T) {
            // D^T[e][r], e = tj*16+n, r = ti*16+m0+j : 8 contiguous bf16 (16B)
            v8bf tmp;
            #pragma unroll
            for (int j = 0; j < 8; ++j) tmp[j] = f2bf(acc[j]);
            *(v8bf*)(D + (tj * 16 + n) * DIM + ti * 16 + m0) = tmp;
        } else {
            bf16* Db = D + (ti * 16) * DIM + tj * 16;
            #pragma unroll
            for (int j = 0; j < 8; ++j) Db[(m0 + j) * DIM + n] = f2bf(acc[j]);
        }
    }
}

__global__ __launch_bounds__(NTHREADS)
void TemporalMessageBank_76836964926247_kernel(
        const int*   __restrict__ idx,
        const float* __restrict__ cur,
        const float* __restrict__ bank,
        const float* __restrict__ Wq, const float* __restrict__ bq,
        const float* __restrict__ Wk, const float* __restrict__ bk,
        const float* __restrict__ Wv, const float* __restrict__ bv,
        const float* __restrict__ Wo, const float* __restrict__ bo,
        const float* __restrict__ gamma, const float* __restrict__ beta,
        float* __restrict__ out)
{
    __shared__ alignas(16) bf16  wLds[DIM * DIM];        // staged Wq/Wk/Wv/Wo
    __shared__ alignas(16) bf16  pastLds[PROWS * DIM];   // gathered bank rows
    __shared__ alignas(16) bf16  kLds[PROWS * DIM];      // K, row-major [r][d]
    __shared__ alignas(16) bf16  vTLds[DIM * PROWS];     // V^T, [d][r]
    __shared__ alignas(16) bf16  qLds[TQ * DIM];
    __shared__ alignas(16) bf16  curLds[TQ * DIM];
    __shared__ alignas(16) bf16  attnFull[TQ * PROWS];   // block-diag softmax wts
    __shared__ alignas(16) bf16  ctxLds[TQ * DIM];
    __shared__ alignas(16) float xLds[TQ * DIM];         // residual sum, fp32
    __shared__ float logitsLds[TQ * BANK];
    __shared__ float redLds[TQ * 8 * 2];
    __shared__ float statLds[TQ * 2];

    const int t    = threadIdx.x;
    const int lane = t & 31;
    const int w    = t >> 5;
    const int n0   = blockIdx.x * TQ;

    // ---- Phase 0: stage Wq; gather past rows (coalesced float4) + cur tile ----
    load_w(Wq, wLds, t);
    {
        const int l = t & 31, g = t >> 5;      // 4 row-groups of 32 lanes
        for (int i = 0; i < PROWS / 4; ++i) {  // one full row per group/iter
            int r    = g + i * 4;
            int node = idx[n0 + (r >> 3)];
            const float* src = bank + ((size_t)node * BANK + (r & 7)) * DIM + l * 4;
            float4 v = *(const float4*)src;
            bf16* d = pastLds + r * DIM + l * 4;
            d[0] = f2bf(v.x); d[1] = f2bf(v.y); d[2] = f2bf(v.z); d[3] = f2bf(v.w);
        }
        for (int i = 0; i < TQ / 4; ++i) {
            int r = g + i * 4;
            const float* src = cur + ((size_t)(n0 + r)) * DIM + l * 4;
            float4 v = *(const float4*)src;
            bf16* d = curLds + r * DIM + l * 4;
            d[0] = f2bf(v.x); d[1] = f2bf(v.y); d[2] = f2bf(v.z); d[3] = f2bf(v.w);
        }
    }
    __syncthreads();

    // ---- Q = cur @ Wq^T + bq ----
    proj_mm<false>(curLds, wLds, bq, qLds, 1, w, lane);
    __syncthreads();
    load_w(Wk, wLds, t);
    __syncthreads();
    // ---- K = past @ Wk^T + bk  (128x128x128 GEMM) ----
    proj_mm<false>(pastLds, wLds, bk, kLds, 8, w, lane);
    __syncthreads();
    load_w(Wv, wLds, t);
    __syncthreads();
    // ---- V^T = (past @ Wv^T + bv)^T  (stored transposed for ctx GEMM) ----
    proj_mm<true>(pastLds, wLds, bv, vTLds, 8, w, lane);
    __syncthreads();

    // ---- stage Wo; logits via WMMA: S = Q @ K^T, keep block diagonal ----
    load_w(Wo, wLds, t);
    {
        // zero attn_full (written sparsely by softmax next phase)
        v8bf z = {};
        *(v8bf*)(attnFull + t * 16)     = z;
        *(v8bf*)(attnFull + t * 16 + 8) = z;

        #pragma unroll
        for (int q = 0; q < 2; ++q) {
            int tj = w * 2 + q;                 // key-row tile
            v8f acc = {};
            #pragma unroll
            for (int c = 0; c < 4; ++c) {
                v16bf a = frag_a(qLds, lane, c);
                v16bf b = frag_b(kLds + tj * 16 * DIM, lane, c);  // B col r = K row r
                acc = __builtin_amdgcn_wmma_f32_16x16x32_bf16(
                          false, a, false, b, (short)0, acc, false, false);
            }
            // lane owns column r = tj*16 + (lane&15), rows m0..m0+7.
            // needed entry: row n = r>>3 (query owning key-row r).
            int r  = tj * 16 + (lane & 15);
            int n  = r >> 3;
            float val = acc[0];
            #pragma unroll
            for (int j = 1; j < 8; ++j) val = (j == (n & 7)) ? acc[j] : val;
            if ((n >> 3) == (lane >> 4))        // n inside this lane-half's rows
                logitsLds[n * BANK + (r & 7)] = val * QK_SCALE;
        }
    }
    __syncthreads();

    // ---- softmax over BANK=8; write bf16 weights into block-diag attn_full ----
    if (t < TQ) {
        float lv[BANK], mx = -1e30f;
        #pragma unroll
        for (int m = 0; m < BANK; ++m) { lv[m] = logitsLds[t * BANK + m]; mx = fmaxf(mx, lv[m]); }
        float s = 0.f;
        #pragma unroll
        for (int m = 0; m < BANK; ++m) { lv[m] = __expf(lv[m] - mx); s += lv[m]; }
        float inv = 1.f / s;
        #pragma unroll
        for (int m = 0; m < BANK; ++m)
            attnFull[t * PROWS + t * BANK + m] = f2bf(lv[m] * inv);
    }
    __syncthreads();

    // ---- ctx = attn_full(16x128) @ V  via WMMA (B-frags from V^T) ----
    {
        #pragma unroll
        for (int q = 0; q < 2; ++q) {
            int tj = w * 2 + q;                 // output-dim tile
            v8f acc = {};
            #pragma unroll
            for (int c = 0; c < 4; ++c) {
                v16bf a = frag_a(attnFull, lane, c);
                v16bf b = frag_b(vTLds + tj * 16 * DIM, lane, c); // B col d = vT row d
                acc = __builtin_amdgcn_wmma_f32_16x16x32_bf16(
                          false, a, false, b, (short)0, acc, false, false);
            }
            int n = lane & 15, m0 = (lane >> 4) << 3;
            bf16* Db = ctxLds + tj * 16;
            #pragma unroll
            for (int j = 0; j < 8; ++j) Db[(m0 + j) * DIM + n] = f2bf(acc[j]);
        }
    }
    __syncthreads();

    // ---- x = cur + ctx @ Wo^T + bo  (fp32 residual fused into tile store) ----
    {
        #pragma unroll
        for (int q = 0; q < 2; ++q) {
            int tj = w * 2 + q;
            float bv = bo[tj * 16 + (lane & 15)];
            v8f acc;
            #pragma unroll
            for (int j = 0; j < 8; ++j) acc[j] = bv;
            const bf16* Wb = wLds + tj * 16 * DIM;
            #pragma unroll
            for (int c = 0; c < 4; ++c) {
                v16bf a = frag_a(ctxLds, lane, c);
                v16bf b = frag_b(Wb, lane, c);
                acc = __builtin_amdgcn_wmma_f32_16x16x32_bf16(
                          false, a, false, b, (short)0, acc, false, false);
            }
            int n = lane & 15, e = tj * 16 + n, m0 = (lane >> 4) << 3;
            #pragma unroll
            for (int j = 0; j < 8; ++j) {
                int m = m0 + j;
                xLds[m * DIM + e] = acc[j] + cur[((size_t)(n0 + m)) * DIM + e];
            }
        }
    }
    __syncthreads();

    // ---- LayerNorm: 8 threads per row, LDS tree reduce ----
    {
        int r = t >> 3, s = t & 7;
        float sm = 0.f, sq = 0.f;
        const float* xr = xLds + r * DIM + s * 16;
        #pragma unroll
        for (int j = 0; j < 16; ++j) { float v = xr[j]; sm += v; sq += v * v; }
        redLds[(r * 8 + s) * 2 + 0] = sm;
        redLds[(r * 8 + s) * 2 + 1] = sq;
    }
    __syncthreads();
    if ((t & 7) == 0) {
        int r = t >> 3;
        float sm = 0.f, sq = 0.f;
        #pragma unroll
        for (int s = 0; s < 8; ++s) { sm += redLds[(r * 8 + s) * 2]; sq += redLds[(r * 8 + s) * 2 + 1]; }
        float mu  = sm * (1.0f / DIM);
        float var = sq * (1.0f / DIM) - mu * mu;
        statLds[r * 2 + 0] = mu;
        statLds[r * 2 + 1] = rsqrtf(var + LN_EPS);
    }
    __syncthreads();
    {
        int r = t >> 3, s = t & 7;
        float mu = statLds[r * 2], rs = statLds[r * 2 + 1];
        float* o = out + ((size_t)(n0 + r)) * DIM;
        const float* xr = xLds + r * DIM;
        #pragma unroll
        for (int j = 0; j < 16; ++j) {
            int d = s * 16 + j;
            o[d] = (xr[d] - mu) * rs * gamma[d] + beta[d];
        }
    }
}

extern "C" void kernel_launch(void* const* d_in, const int* in_sizes, int n_in,
                              void* d_out, int out_size, void* d_ws, size_t ws_size,
                              hipStream_t stream) {
    const int*   idx   = (const int*)  d_in[0];
    const float* cur   = (const float*)d_in[1];
    const float* bank  = (const float*)d_in[2];
    const float* Wq    = (const float*)d_in[3];
    const float* bq    = (const float*)d_in[4];
    const float* Wk    = (const float*)d_in[5];
    const float* bk    = (const float*)d_in[6];
    const float* Wv    = (const float*)d_in[7];
    const float* bv    = (const float*)d_in[8];
    const float* Wo    = (const float*)d_in[9];
    const float* bo    = (const float*)d_in[10];
    const float* gamma = (const float*)d_in[11];
    const float* beta  = (const float*)d_in[12];
    float* out = (float*)d_out;

    const int nreads = in_sizes[0];            // 65536
    dim3 grid(nreads / TQ), block(NTHREADS);
    TemporalMessageBank_76836964926247_kernel<<<grid, block, 0, stream>>>(
        idx, cur, bank, Wq, bq, Wk, bk, Wv, bv, Wo, bo, gamma, beta, out);
}
